// EDTransformer_71554155151930
// MI455X (gfx1250) — matrix operand order repaired and checked by reading, not compiled, not run on hardware
//
#include <hip/hip_runtime.h>
#include <hip/hip_bf16.h>
#include <math.h>

// ---------------------------------------------------------------------------
// Types for CDNA5 WMMA (wave32): v16bf A/B fragments, v8f accumulator.
// ---------------------------------------------------------------------------
typedef __attribute__((ext_vector_type(16))) __bf16 v16bf;
typedef __attribute__((ext_vector_type(8)))  __bf16 v8bf;
typedef __attribute__((ext_vector_type(8)))  float  v8f;

union FragBF { v16bf v; v8bf h[2]; };

#define BM   128
#define BKK  32
#define LSTR 40   // LDS row stride (elements) = 80B: 16B aligned, spreads banks

enum { F_BIAS = 1, F_RELU = 2, F_RES = 4, F_TRANSB = 8, F_NT = 16 };

struct GemmP {
  const float* A; const float* B; const float* bias; const float* res; float* C;
  int M, N, K, lda, ldb, ldc;
  long long aO, aI, bO, bI, cO, cI, biasO, biasI;  // outer/inner batch strides
  int innerCnt;          // batch index split: outer = z/innerCnt, inner = z%innerCnt
  float alpha;
  int flags;
};

// ---------------------------------------------------------------------------
// Generalized batched GEMM: C = alpha*(A x B) [+bias][+res][relu]
//   A: MxK f32 (row major, lda), B: KxN f32 (row major, ldb) or NxK if TB.
//   f32 -> bf16 conversion while staging into LDS; core is
//   v_wmma_f32_16x16x32_bf16 with double-buffered LDS ping-pong and a
//   branch-free steady-state loop (last iteration peeled).
//   All dims used here are multiples of the tile sizes (no bounds checks).
// ---------------------------------------------------------------------------
template <int BN_, bool TB>
__global__ __launch_bounds__(256) void gemm_bf16_wmma(GemmP p) {
  constexpr int NT = BN_ / 32;   // 16-col C tiles per wave (waveN covers BN_/2 cols)
  constexpr int NP = BN_ / 32;   // B-staging passes (float4s per thread)

  __shared__ __align__(16) __bf16 As[2][BM * LSTR];
  __shared__ __align__(16) __bf16 Bs[2][BN_ * LSTR];

  const int tid  = threadIdx.x;
  const int lane = tid & 31;
  const int wave = tid >> 5;
  const int waveM = wave & 3;      // 4 waves along M (32 rows each)
  const int waveN = wave >> 2;     // 2 waves along N (BN_/2 cols each)

  const int bz   = blockIdx.z;
  const int obat = bz / p.innerCnt;
  const int ibat = bz % p.innerCnt;
  const long long offA    = (long long)obat * p.aO    + (long long)ibat * p.aI;
  const long long offB    = (long long)obat * p.bO    + (long long)ibat * p.bI;
  const long long offC    = (long long)obat * p.cO    + (long long)ibat * p.cI;
  const long long offBias = (long long)obat * p.biasO + (long long)ibat * p.biasI;

  const int n0 = blockIdx.x * BN_;
  const float* Ag = p.A + offA + (long long)blockIdx.y * BM * p.lda;
  const float* Bg = p.B + offB;
  float*       Cg = p.C + offC + (long long)blockIdx.y * BM * p.ldc + n0;
  const float* Rg = (p.flags & F_RES)
                  ? (p.res + offC + (long long)blockIdx.y * BM * p.ldc + n0) : nullptr;

  // per-thread staging coordinates
  const int ac4 = (tid & 7) * 4;            // A / B(trans): k offset within 32
  const int ar0 = tid >> 3;                 // A / B(trans): row base (stride 32)
  const int bn4 = (tid & (BN_/4 - 1)) * 4;  // B non-trans: n offset
  const int bk0 = tid / (BN_/4);            // B non-trans: k base
  constexpr int BKS = 1024 / BN_;           // B non-trans: k step per pass

  float4 fa[4], fb[NP];

  // ---- gather one K-tile's worth of global data into registers ----
  auto loadA = [&](int k0) {
    #pragma unroll
    for (int pp = 0; pp < 4; ++pp)
      fa[pp] = *(const float4*)(Ag + (long long)(ar0 + pp * 32) * p.lda + k0 + ac4);
  };
  auto loadB = [&](int k0) {
    if (TB) {      // element (k,n) at B + n*ldb + k
      #pragma unroll
      for (int pp = 0; pp < NP; ++pp)
        fb[pp] = *(const float4*)(Bg + (long long)(n0 + ar0 + pp * 32) * p.ldb + k0 + ac4);
    } else {       // element (k,n) at B + k*ldb + n
      #pragma unroll
      for (int pp = 0; pp < NP; ++pp)
        fb[pp] = *(const float4*)(Bg + (long long)(k0 + bk0 + pp * BKS) * p.ldb + n0 + bn4);
    }
  };
  // ---- convert to bf16 and store into LDS buffer ----
  auto storeA = [&](__bf16* buf) {
    #pragma unroll
    for (int pp = 0; pp < 4; ++pp) {
      __bf16* d = &buf[(ar0 + pp * 32) * LSTR + ac4];
      d[0] = (__bf16)fa[pp].x; d[1] = (__bf16)fa[pp].y;
      d[2] = (__bf16)fa[pp].z; d[3] = (__bf16)fa[pp].w;
    }
  };
  auto storeB = [&](__bf16* buf) {
    if (TB) {
      #pragma unroll
      for (int pp = 0; pp < NP; ++pp) {
        __bf16* d = &buf[(ar0 + pp * 32) * LSTR + ac4];
        d[0] = (__bf16)fb[pp].x; d[1] = (__bf16)fb[pp].y;
        d[2] = (__bf16)fb[pp].z; d[3] = (__bf16)fb[pp].w;
      }
    } else {
      #pragma unroll
      for (int pp = 0; pp < NP; ++pp) {
        const int kk = bk0 + pp * BKS;
        buf[(bn4 + 0) * LSTR + kk] = (__bf16)fb[pp].x;
        buf[(bn4 + 1) * LSTR + kk] = (__bf16)fb[pp].y;
        buf[(bn4 + 2) * LSTR + kk] = (__bf16)fb[pp].z;
        buf[(bn4 + 3) * LSTR + kk] = (__bf16)fb[pp].w;
      }
    }
  };

  v8f acc[2][NT] = {};

  // fragments per ISA 16-bit layouts:
  //   A 16x32: lane l -> row M=l&15, K = (l>>4)*8 .. +7 and +16 .. +23
  //   B 32x16: lane l -> col N=l&15, K = (l>>4)*8 .. +7 and +16 .. +23
  const int am   = lane & 15;
  const int kb8  = (lane >> 4) * 8;
  const int kb16 = (lane >> 4) * 16;

  auto compute = [&](int buf) {
    const __bf16* Ab = As[buf];
    const __bf16* Bb = Bs[buf];
    FragBF afr[2], bfr[NT];
    #pragma unroll
    for (int mt = 0; mt < 2; ++mt) {
      const int row = waveM * 32 + mt * 16 + am;
      afr[mt].h[0] = *(const v8bf*)&Ab[row * LSTR + kb8];
      afr[mt].h[1] = *(const v8bf*)&Ab[row * LSTR + kb8 + 16];
    }
    #pragma unroll
    for (int nt = 0; nt < NT; ++nt) {
      const int nr = waveN * (BN_/2) + nt * 16 + am;
      bfr[nt].h[0] = *(const v8bf*)&Bb[nr * LSTR + kb8];
      bfr[nt].h[1] = *(const v8bf*)&Bb[nr * LSTR + kb8 + 16];
    }
    #pragma unroll
    for (int mt = 0; mt < 2; ++mt)
      #pragma unroll
      for (int nt = 0; nt < NT; ++nt)
        acc[mt][nt] = __builtin_amdgcn_wmma_f32_16x16x32_bf16(
            false, afr[mt].v, false, bfr[nt].v, (short)0, acc[mt][nt], false, false);
  };

  // ---- prologue: stage first tile ----
  loadA(0); loadB(0);
  storeA(As[0]); storeB(Bs[0]);
  __syncthreads();

  const int iters = p.K / BKK;
  // ---- branch-free steady state: load k+1, compute k, store k+1, barrier ----
  for (int it = 0; it < iters - 1; ++it) {
    const int cur = it & 1;
    const int nxt = cur ^ 1;
    loadA((it + 1) * BKK); loadB((it + 1) * BKK);
    compute(cur);
    storeA(As[nxt]); storeB(Bs[nxt]);
    __syncthreads();
  }
  // ---- peeled final tile: compute only ----
  compute((iters - 1) & 1);

  // ---- epilogue: C 16x16 layout: VGPR r -> (M=r, N=lane) / (M=8+r, N=lane-16)
  const int colL = lane & 15;
  const int rb   = (lane >> 4) * 8;
  #pragma unroll
  for (int mt = 0; mt < 2; ++mt) {
    #pragma unroll
    for (int nt = 0; nt < NT; ++nt) {
      const int col = waveN * (BN_/2) + nt * 16 + colL;
      float bv = 0.f;
      if (p.flags & F_BIAS) bv = p.bias[offBias + n0 + col];
      #pragma unroll
      for (int r = 0; r < 8; ++r) {
        const int row = waveM * 32 + mt * 16 + rb + r;
        float vv = acc[mt][nt][r] * p.alpha + bv;
        if (p.flags & F_RES) vv += Rg[(long long)row * p.ldc + col];
        if (p.flags & F_RELU) vv = fmaxf(vv, 0.f);
        if (p.flags & F_NT) __builtin_nontemporal_store(vv, &Cg[(long long)row * p.ldc + col]);
        else                Cg[(long long)row * p.ldc + col] = vv;
      }
    }
  }
}

// ---------------------------------------------------------------------------
// Row softmax over [rows x T] with optional causal mask (col <= row%Srows).
// ---------------------------------------------------------------------------
__global__ __launch_bounds__(256) void softmax_rows(float* sc, int T, int Srows, int causal) {
  __shared__ float red[256];
  const long long row = blockIdx.x;
  float* p = sc + row * (long long)T;
  const int s   = (int)(row % Srows);
  const int tid = threadIdx.x;
  const int lim = causal ? (s + 1) : T;

  float m = -3.4e38f;
  for (int t = tid; t < lim; t += 256) m = fmaxf(m, p[t]);
  red[tid] = m; __syncthreads();
  for (int o = 128; o > 0; o >>= 1) { if (tid < o) red[tid] = fmaxf(red[tid], red[tid + o]); __syncthreads(); }
  m = red[0]; __syncthreads();

  float sum = 0.f;
  for (int t = tid; t < T; t += 256) {
    float v = (t < lim) ? __expf(p[t] - m) : 0.f;
    p[t] = v; sum += v;
  }
  red[tid] = sum; __syncthreads();
  for (int o = 128; o > 0; o >>= 1) { if (tid < o) red[tid] += red[tid + o]; __syncthreads(); }
  const float inv = 1.f / red[0];
  __syncthreads();
  for (int t = tid; t < T; t += 256) p[t] *= inv;
}

// ---------------------------------------------------------------------------
// In-place LayerNorm over rows of length Dn.
// ---------------------------------------------------------------------------
__global__ __launch_bounds__(256) void layernorm_rows(float* x, const float* sc,
                                                      const float* of, int Dn) {
  __shared__ float r1[256], r2[256];
  float* p = x + (long long)blockIdx.x * Dn;
  const int tid = threadIdx.x;
  float s = 0.f, q = 0.f;
  for (int d = tid; d < Dn; d += 256) { float v = p[d]; s += v; q += v * v; }
  r1[tid] = s; r2[tid] = q; __syncthreads();
  for (int o = 128; o > 0; o >>= 1) {
    if (tid < o) { r1[tid] += r1[tid + o]; r2[tid] += r2[tid + o]; }
    __syncthreads();
  }
  const float mean = r1[0] / Dn;
  const float var  = r2[0] / Dn - mean * mean;
  const float inv  = rsqrtf(var + 1e-5f);
  for (int d = tid; d < Dn; d += 256) p[d] = (p[d] - mean) * inv * sc[d] + of[d];
}

// ---------------------------------------------------------------------------
// Embedding: out[b,s,:] = tok_emb[ids[b,s],:] + pos_emb[s,:]
// ---------------------------------------------------------------------------
__global__ __launch_bounds__(256) void embed_tokens(const int* ids, const float* tok,
                                                    const float* pos, float* out,
                                                    int Sdim, int Dn, long long total) {
  long long i = (long long)blockIdx.x * 256 + threadIdx.x;
  if (i >= total) return;
  const int d = (int)(i % Dn);
  const long long sd = i / Dn;
  const int s  = (int)(sd % Sdim);
  const int id = ids[sd];
  out[i] = tok[(long long)id * Dn + d] + pos[(long long)s * Dn + d];
}

// ---------------------------------------------------------------------------
// Host-side orchestration
// ---------------------------------------------------------------------------
static const int Bc = 2, Sc = 1024, Dc = 1024, Hc = 16, HDc = 64, Vc = 32000, NLc = 4;

struct AttnW { const float *Wk, *Wo, *Wq, *Wv, *bk, *bo, *bq, *bv; };
struct FfnW  { const float *W1, *W2, *b1, *b2; };
struct DecL  { AttnW a1, a2; FfnW f; const float *ln1o,*ln1s,*ln2o,*ln2s,*ln3o,*ln3s; };
struct EncL  { AttnW a;  FfnW f; const float *ln1o,*ln1s,*ln2o,*ln2s; };

static AttnW readAttn(void* const* d, int& i) {
  AttnW w;                               // sorted keys: Wk,Wo,Wq,Wv,bk,bo,bq,bv
  w.Wk = (const float*)d[i++]; w.Wo = (const float*)d[i++];
  w.Wq = (const float*)d[i++]; w.Wv = (const float*)d[i++];
  w.bk = (const float*)d[i++]; w.bo = (const float*)d[i++];
  w.bq = (const float*)d[i++]; w.bv = (const float*)d[i++];
  return w;
}
static FfnW readFfn(void* const* d, int& i) {
  FfnW w;                                // sorted keys: W1,W2,b1,b2
  w.W1 = (const float*)d[i++]; w.W2 = (const float*)d[i++];
  w.b1 = (const float*)d[i++]; w.b2 = (const float*)d[i++];
  return w;
}

static GemmP mkP(const float* A, const float* B, const float* bias, const float* res,
                 float* C, int M, int N, int K, int lda, int ldb, int ldc,
                 float alpha, int flags, int innerCnt,
                 long long aO, long long aI, long long bO, long long bI,
                 long long cO, long long cI, long long biasO, long long biasI) {
  GemmP p; p.A = A; p.B = B; p.bias = bias; p.res = res; p.C = C;
  p.M = M; p.N = N; p.K = K; p.lda = lda; p.ldb = ldb; p.ldc = ldc;
  p.aO = aO; p.aI = aI; p.bO = bO; p.bI = bI; p.cO = cO; p.cI = cI;
  p.biasO = biasO; p.biasI = biasI;
  p.innerCnt = innerCnt > 0 ? innerCnt : 1; p.alpha = alpha; p.flags = flags;
  return p;
}

static void launch_gemm(hipStream_t st, const GemmP& p, int nbatch) {
  const bool tb = (p.flags & F_TRANSB) != 0;
  if ((p.N % 128) == 0) {
    dim3 g(p.N / 128, p.M / BM, nbatch);
    if (tb) gemm_bf16_wmma<128, true ><<<g, dim3(256), 0, st>>>(p);
    else    gemm_bf16_wmma<128, false><<<g, dim3(256), 0, st>>>(p);
  } else {
    dim3 g(p.N / 64, p.M / BM, nbatch);
    if (tb) gemm_bf16_wmma<64, true ><<<g, dim3(256), 0, st>>>(p);
    else    gemm_bf16_wmma<64, false><<<g, dim3(256), 0, st>>>(p);
  }
}

static void run_mha(hipStream_t st, const float* xin, const float* kvin, float* xio,
                    float* q, float* k, float* v, float* y, float* sc,
                    const AttnW& w, int causal) {
  const int BS = Bc * Sc;
  const long long SD = (long long)Sc * Dc;
  const long long SS = (long long)Sc * Sc;
  // per-head Q/K/V projections (block diagonal) -- batched over heads
  launch_gemm(st, mkP(xin,  w.Wq, w.bq, nullptr, q, BS, HDc, HDc, Dc, HDc, Dc,
                      1.f, F_BIAS, Hc, 0, HDc, 0, (long long)HDc * HDc, 0, HDc, 0, HDc), Hc);
  launch_gemm(st, mkP(kvin, w.Wk, w.bk, nullptr, k, BS, HDc, HDc, Dc, HDc, Dc,
                      1.f, F_BIAS, Hc, 0, HDc, 0, (long long)HDc * HDc, 0, HDc, 0, HDc), Hc);
  launch_gemm(st, mkP(kvin, w.Wv, w.bv, nullptr, v, BS, HDc, HDc, Dc, HDc, Dc,
                      1.f, F_BIAS, Hc, 0, HDc, 0, (long long)HDc * HDc, 0, HDc, 0, HDc), Hc);
  // scores = (Q K^T)/sqrt(HD); batch over (b,h)
  launch_gemm(st, mkP(q, k, nullptr, nullptr, sc, Sc, Sc, HDc, Dc, Dc, Sc,
                      0.125f, F_TRANSB, Hc, SD, HDc, SD, HDc,
                      (long long)Hc * SS, SS, 0, 0), Bc * Hc);
  hipLaunchKernelGGL(softmax_rows, dim3(Bc * Hc * Sc), dim3(256), 0, st, sc, Sc, Sc, causal);
  // y = P V
  launch_gemm(st, mkP(sc, v, nullptr, nullptr, y, Sc, HDc, Sc, Sc, Dc, Dc,
                      1.f, 0, Hc, (long long)Hc * SS, SS, SD, HDc, SD, HDc, 0, 0), Bc * Hc);
  // out = xin + y Wo + bo
  launch_gemm(st, mkP(y, w.Wo, w.bo, xin, xio, BS, Dc, Dc, Dc, Dc, Dc,
                      1.f, F_BIAS | F_RES, 1, 0, 0, 0, 0, 0, 0, 0, 0), 1);
}

static void run_ffn(hipStream_t st, float* x, float* t, const FfnW& w) {
  const int BS = Bc * Sc;
  launch_gemm(st, mkP(x, w.W1, w.b1, nullptr, t, BS, Dc, Dc, Dc, Dc, Dc,
                      1.f, F_BIAS | F_RELU, 1, 0, 0, 0, 0, 0, 0, 0, 0), 1);
  launch_gemm(st, mkP(t, w.W2, w.b2, x, x, BS, Dc, Dc, Dc, Dc, Dc,
                      1.f, F_BIAS | F_RES, 1, 0, 0, 0, 0, 0, 0, 0, 0), 1);
}

extern "C" void kernel_launch(void* const* d_in, const int* in_sizes, int n_in,
                              void* d_out, int out_size, void* d_ws, size_t ws_size,
                              hipStream_t stream) {
  (void)in_sizes; (void)n_in; (void)out_size; (void)ws_size;
  // pytree order (sorted dict keys): X, Z, params{Wu, dec[0..3], enc[0..3], pos_emb, tok_emb}
  int i = 0;
  const int* Xids = (const int*)d_in[i++];
  const int* Zids = (const int*)d_in[i++];
  const float* Wu = (const float*)d_in[i++];
  DecL dec[NLc];
  for (int l = 0; l < NLc; ++l) {           // attn1, attn2, ffn, ln1o..ln3s
    dec[l].a1 = readAttn(d_in, i);
    dec[l].a2 = readAttn(d_in, i);
    dec[l].f  = readFfn(d_in, i);
    dec[l].ln1o = (const float*)d_in[i++]; dec[l].ln1s = (const float*)d_in[i++];
    dec[l].ln2o = (const float*)d_in[i++]; dec[l].ln2s = (const float*)d_in[i++];
    dec[l].ln3o = (const float*)d_in[i++]; dec[l].ln3s = (const float*)d_in[i++];
  }
  EncL enc[NLc];
  for (int l = 0; l < NLc; ++l) {           // attn, ffn, ln1o..ln2s
    enc[l].a = readAttn(d_in, i);
    enc[l].f = readFfn(d_in, i);
    enc[l].ln1o = (const float*)d_in[i++]; enc[l].ln1s = (const float*)d_in[i++];
    enc[l].ln2o = (const float*)d_in[i++]; enc[l].ln2s = (const float*)d_in[i++];
  }
  const float* pos = (const float*)d_in[i++];
  const float* tok = (const float*)d_in[i++];

  // workspace carve-up (floats): z, x, q, k, v, y, t (B*S*D each) + scores (B*H*S*S)
  const size_t BSD = (size_t)Bc * Sc * Dc;
  float* ws = (float*)d_ws;
  float* zb = ws;            float* xb = zb + BSD;
  float* q  = xb + BSD;      float* k  = q + BSD;
  float* v  = k + BSD;       float* y  = v + BSD;
  float* t  = y + BSD;       float* sc = t + BSD;

  const long long tot = (long long)BSD;
  const int eg = (int)((tot + 255) / 256);

  // ---- encoder ----
  hipLaunchKernelGGL(embed_tokens, dim3(eg), dim3(256), 0, stream,
                     Zids, tok, pos, zb, Sc, Dc, tot);
  for (int l = 0; l < NLc; ++l) {
    run_mha(stream, zb, zb, zb, q, k, v, y, sc, enc[l].a, 0);
    hipLaunchKernelGGL(layernorm_rows, dim3(Bc * Sc), dim3(256), 0, stream,
                       zb, enc[l].ln1s, enc[l].ln1o, Dc);
    run_ffn(stream, zb, t, enc[l].f);
    hipLaunchKernelGGL(layernorm_rows, dim3(Bc * Sc), dim3(256), 0, stream,
                       zb, enc[l].ln2s, enc[l].ln2o, Dc);
  }

  // ---- decoder ----
  hipLaunchKernelGGL(embed_tokens, dim3(eg), dim3(256), 0, stream,
                     Xids, tok, pos, xb, Sc, Dc, tot);
  for (int l = 0; l < NLc; ++l) {
    run_mha(stream, xb, xb, xb, q, k, v, y, sc, dec[l].a1, 1);   // causal self-attn
    hipLaunchKernelGGL(layernorm_rows, dim3(Bc * Sc), dim3(256), 0, stream,
                       xb, dec[l].ln1s, dec[l].ln1o, Dc);
    run_mha(stream, xb, zb, xb, q, k, v, y, sc, dec[l].a2, 0);   // cross-attn
    hipLaunchKernelGGL(layernorm_rows, dim3(Bc * Sc), dim3(256), 0, stream,
                       xb, dec[l].ln2s, dec[l].ln2o, Dc);
    run_ffn(stream, xb, t, dec[l].f);
    hipLaunchKernelGGL(layernorm_rows, dim3(Bc * Sc), dim3(256), 0, stream,
                       xb, dec[l].ln3s, dec[l].ln3o, Dc);
  }

  // ---- logits: [B*S, D] x [D, V] -> d_out (streamed past L2) ----
  launch_gemm(stream, mkP(xb, Wu, nullptr, nullptr, (float*)d_out,
                          Bc * Sc, Vc, Dc, Dc, Vc, Vc,
                          1.f, F_NT, 1, 0, 0, 0, 0, 0, 0, 0, 0), 1);
}
